// MultiHeadAttention_69114613727310
// MI455X (gfx1250) — compile-verified, matrix operand order checked by reading, not run on hardware
//
#include <hip/hip_runtime.h>
#include <hip/hip_bf16.h>

#define EMB   1024
#define NHEAD 16
#define HDIM  64
#define SEQ   2048
#define BATCH 4

typedef __bf16 bf16_t;
typedef __attribute__((ext_vector_type(16))) __bf16 v16bf;
typedef __attribute__((ext_vector_type(8)))  __bf16 v8bf;
typedef __attribute__((ext_vector_type(8)))  float  v8f;

__device__ __forceinline__ bf16_t f2bf(float f) {
    union { float f; unsigned u; } x; x.f = f;
    unsigned r = (x.u + 0x7FFFu + ((x.u >> 16) & 1u)) >> 16;
    union { unsigned short s; bf16_t b; } y; y.s = (unsigned short)r;
    return y.b;
}

// ---- DPP 16-lane butterfly reductions (no LDS, no ds_bpermute) ----
template <int CTRL>
__device__ __forceinline__ float dpp_movf(float x) {
    int r = __builtin_amdgcn_update_dpp(0, __builtin_bit_cast(int, x),
                                        CTRL, 0xF, 0xF, true);
    return __builtin_bit_cast(float, r);
}
__device__ __forceinline__ float red16_max(float x) {
    x = fmaxf(x, dpp_movf<0xB1>(x));    // quad_perm(1,0,3,2)  : xor 1
    x = fmaxf(x, dpp_movf<0x4E>(x));    // quad_perm(2,3,0,1)  : xor 2
    x = fmaxf(x, dpp_movf<0x141>(x));   // row_half_mirror     : merge quads in 8
    x = fmaxf(x, dpp_movf<0x140>(x));   // row_mirror          : merge 8s in 16
    return x;
}
__device__ __forceinline__ float red16_sum(float x) {
    x += dpp_movf<0xB1>(x);
    x += dpp_movf<0x4E>(x);
    x += dpp_movf<0x141>(x);
    x += dpp_movf<0x140>(x);
    return x;
}

// Load a 16x32 bf16 WMMA fragment (A layout; also valid as B when the source
// matrix is stored N-major with K contiguous, i.e. "row = n, col = k").
// ISA 16-bit A layout: lane L -> row L%16, half = L/16;
// halfs [0..7]  = k = half*8 + 0..7   (one b128 load)
// halfs [8..15] = same + 16           (one b128 load)
__device__ __forceinline__ v16bf load_frag(const bf16_t* p, int row0, int k0, int ld) {
    int lane = threadIdx.x & 31;
    int r = lane & 15, hf = lane >> 4;
    const bf16_t* base = p + (size_t)(row0 + r) * ld + k0 + hf * 8;
    v8bf lo = *(const v8bf*)(base);
    v8bf hi = *(const v8bf*)(base + 16);
    return __builtin_shufflevector(lo, hi, 0,1,2,3,4,5,6,7,8,9,10,11,12,13,14,15);
}

__device__ __forceinline__ void prefetch_frag(const bf16_t* p, int row0, int k0, int ld) {
    int lane = threadIdx.x & 31;
    int r = lane & 15, hf = lane >> 4;
    __builtin_prefetch(p + (size_t)(row0 + r) * ld + k0 + hf * 8, 0, 3);
}

__device__ __forceinline__ v8f wmma_bf16(v16bf a, v16bf b, v8f c) {
    return __builtin_amdgcn_wmma_f32_16x16x32_bf16(false, a, false, b, (short)0, c, false, false);
}

// ---------------------------------------------------------------- converts
__global__ __launch_bounds__(256) void cvt_f32_bf16(const float* __restrict__ in,
                                                    bf16_t* __restrict__ out, int n) {
    int i = blockIdx.x * 256 + threadIdx.x;
    if (i < n) out[i] = f2bf(in[i]);
}

// V (B*S, E) -> Vt (B*H*D, S)  so that P@V's B-fragment has s' contiguous.
__global__ __launch_bounds__(256) void transpose_v(const bf16_t* __restrict__ V,
                                                   bf16_t* __restrict__ Vt, int n) {
    int i = blockIdx.x * 256 + threadIdx.x;
    if (i >= n) return;
    int d = i & 63;
    int h = (i >> 6) & (NHEAD - 1);
    int s = (i >> 10) & (SEQ - 1);
    int b = i >> 21;
    Vt[((size_t)((b * NHEAD + h) * HDIM + d)) * SEQ + s] = V[i];
}

// ---------------------------------------------------------------- GEMM
// Y[M,N] = X[M,K] @ W[N,K]^T + bias[N].  One wave computes a 32x64 tile:
// 2 A-frags + 4 B-frags -> 8 WMMAs per K-step (~21 flop/byte from cache).
template <bool BF16OUT>
__global__ __launch_bounds__(256) void gemm_bias(const bf16_t* __restrict__ X,
                                                 const bf16_t* __restrict__ W,
                                                 const float* __restrict__ bias,
                                                 void* __restrict__ Yv,
                                                 int M, int N, int K) {
    int wave = threadIdx.x >> 5;
    int gw = blockIdx.x * 8 + wave;
    int ntiles = N >> 6;
    int mt = gw / ntiles, nt = gw % ntiles;
    if (mt * 32 >= M) return;                 // wave-uniform: EXEC stays all-ones
    v8f acc[2][4] = {};
    for (int k0 = 0; k0 < K; k0 += 32) {
        if (k0 + 32 < K) {                    // uniform; lowers to global_prefetch_b8
            prefetch_frag(X, mt * 32,      k0 + 32, K);
            prefetch_frag(X, mt * 32 + 16, k0 + 32, K);
        }
        v16bf a0 = load_frag(X, mt * 32,      k0, K);
        v16bf a1 = load_frag(X, mt * 32 + 16, k0, K);
#pragma unroll
        for (int j = 0; j < 4; ++j) {
            v16bf b = load_frag(W, nt * 64 + j * 16, k0, K);
            acc[0][j] = wmma_bf16(a0, b, acc[0][j]);
            acc[1][j] = wmma_bf16(a1, b, acc[1][j]);
        }
    }
    int lane = threadIdx.x & 31, nn = lane & 15, hf = lane >> 4;
#pragma unroll
    for (int mi = 0; mi < 2; ++mi) {
#pragma unroll
        for (int j = 0; j < 4; ++j) {
            int col = nt * 64 + j * 16 + nn;
            float bv = bias[col];
#pragma unroll
            for (int v = 0; v < 8; ++v) {
                int row = mt * 32 + mi * 16 + v + 8 * hf;   // C layout: m = v + 8*half
                float val = acc[mi][j][v] + bv;
                if (BF16OUT) ((bf16_t*)Yv)[(size_t)row * N + col] = f2bf(val);
                else         ((float*)Yv)[(size_t)row * N + col]  = val;
            }
        }
    }
}

// ---------------------------------------------------------------- attention
// One 64-key flash block (amortizes softmax fixed costs over 2x keys).
// MASKED=false skips all causal compares/selects.
template <bool MASKED>
__device__ __forceinline__ void attn_block(int kb, int q0, int hcol, int nn, int hf,
                                           const bf16_t* __restrict__ Kb,
                                           const bf16_t* __restrict__ Vh,
                                           bf16_t* __restrict__ ldsP,
                                           v16bf aq0, v16bf aq1,
                                           v8f& o0, v8f& o1, v8f& o2, v8f& o3,
                                           float (&mrow)[8], float (&lrow)[8]) {
    const float scale = 0.125f;                 // 1/sqrt(64)
    v8f s[4] = {};
#pragma unroll
    for (int t = 0; t < 4; ++t) {
        s[t] = wmma_bf16(aq0, load_frag(Kb, kb + t * 16, hcol,      EMB), s[t]);
        s[t] = wmma_bf16(aq1, load_frag(Kb, kb + t * 16, hcol + 32, EMB), s[t]);
    }
#pragma unroll
    for (int v = 0; v < 8; ++v) {
        int m = v + 8 * hf;                     // row within tile (C layout)
        float x[4];
#pragma unroll
        for (int t = 0; t < 4; ++t) {
            x[t] = s[t][v] * scale;
            if (MASKED) {
                int qi = q0 + m;
                x[t] = (kb + t * 16 + nn <= qi) ? x[t] : -1e30f;
            }
        }
        float mx = red16_max(fmaxf(fmaxf(x[0], x[1]), fmaxf(x[2], x[3])));
        float nm = fmaxf(mrow[v], mx);
        float p0 = __expf(x[0] - nm);
        float p1 = __expf(x[1] - nm);
        float p2 = __expf(x[2] - nm);
        float p3 = __expf(x[3] - nm);
        float rs = red16_sum((p0 + p1) + (p2 + p3));
        float corr = __expf(mrow[v] - nm);
        lrow[v] = lrow[v] * corr + rs;
        o0[v] *= corr; o1[v] *= corr; o2[v] *= corr; o3[v] *= corr;
        mrow[v] = nm;
        ldsP[m * 64 + nn]      = f2bf(p0);      // C-layout -> LDS (16x64 tile)
        ldsP[m * 64 + 16 + nn] = f2bf(p1);
        ldsP[m * 64 + 32 + nn] = f2bf(p2);
        ldsP[m * 64 + 48 + nn] = f2bf(p3);
    }
    asm volatile("s_wait_dscnt 0" ::: "memory");    // LDS writes -> reads (same wave)
    v16bf ap0 = load_frag(ldsP, 0,  0, 64);         // P keys kb   .. kb+31 (A layout)
    v16bf ap1 = load_frag(ldsP, 0, 32, 64);         // P keys kb+32.. kb+63
    o0 = wmma_bf16(ap0, load_frag(Vh,  0, kb,      SEQ), o0);
    o1 = wmma_bf16(ap0, load_frag(Vh, 16, kb,      SEQ), o1);
    o2 = wmma_bf16(ap0, load_frag(Vh, 32, kb,      SEQ), o2);
    o3 = wmma_bf16(ap0, load_frag(Vh, 48, kb,      SEQ), o3);
    o0 = wmma_bf16(ap1, load_frag(Vh,  0, kb + 32, SEQ), o0);
    o1 = wmma_bf16(ap1, load_frag(Vh, 16, kb + 32, SEQ), o1);
    o2 = wmma_bf16(ap1, load_frag(Vh, 32, kb + 32, SEQ), o2);
    o3 = wmma_bf16(ap1, load_frag(Vh, 48, kb + 32, SEQ), o3);
}

// One wave per (b, h, 16-query tile); all waves in a block share (b, qt)
// (uniform trip count), heavy q-tiles scheduled first.
__global__ __launch_bounds__(256) void attn_fwd(const bf16_t* __restrict__ Q,
                                                const bf16_t* __restrict__ Kmat,
                                                const bf16_t* __restrict__ Vt,
                                                bf16_t* __restrict__ O) {
    __shared__ __align__(16) bf16_t lds_p[8][16 * 64];   // per-wave P tile (16x64)
    int wave = threadIdx.x >> 5;
    int gw = blockIdx.x * 8 + wave;
    int h   = gw & (NHEAD - 1);
    int tmp = gw >> 4;
    int qt  = (SEQ / 16 - 1) - (tmp & (SEQ / 16 - 1));   // descending work size
    int b   = tmp >> 7;
    int lane = threadIdx.x & 31, nn = lane & 15, hf = lane >> 4;

    const bf16_t* Qb = Q    + (size_t)b * SEQ * EMB;
    const bf16_t* Kb = Kmat + (size_t)b * SEQ * EMB;
    const bf16_t* Vh = Vt   + (size_t)(b * NHEAD + h) * HDIM * SEQ;
    bf16_t* ldsP = &lds_p[wave][0];
    int q0 = qt * 16;
    int hcol = h * 64;

    v16bf aq0 = load_frag(Qb, q0, hcol,      EMB);
    v16bf aq1 = load_frag(Qb, q0, hcol + 32, EMB);

    v8f o0 = {}, o1 = {}, o2 = {}, o3 = {};
    float mrow[8], lrow[8];
#pragma unroll
    for (int v = 0; v < 8; ++v) { mrow[v] = -1e30f; lrow[v] = 0.f; }

    // 64-key blocks fully below the diagonal need no mask: kb+63 <= q0.
    // Key rows loaded never exceed SEQ-1 (kb <= 64*floor((q0+15)/64)).
    int kend = q0 + 16;
    int nomask_end = (q0 >= 63) ? ((((q0 - 63) >> 6) << 6) + 64) : 0;
    for (int kb = 0; kb < nomask_end; kb += 64)
        attn_block<false>(kb, q0, hcol, nn, hf, Kb, Vh, ldsP,
                          aq0, aq1, o0, o1, o2, o3, mrow, lrow);
    for (int kb = nomask_end; kb < kend; kb += 64)
        attn_block<true>(kb, q0, hcol, nn, hf, Kb, Vh, ldsP,
                         aq0, aq1, o0, o1, o2, o3, mrow, lrow);

    bf16_t* Ob = O + (size_t)b * SEQ * EMB;
#pragma unroll
    for (int v = 0; v < 8; ++v) {
        float inv = 1.0f / lrow[v];
        size_t base = (size_t)(q0 + v + 8 * hf) * EMB + hcol + nn;
        Ob[base +  0] = f2bf(o0[v] * inv);
        Ob[base + 16] = f2bf(o1[v] * inv);
        Ob[base + 32] = f2bf(o2[v] * inv);
        Ob[base + 48] = f2bf(o3[v] * inv);
    }
}

// ---------------------------------------------------------------- launch
extern "C" void kernel_launch(void* const* d_in, const int* in_sizes, int n_in,
                              void* d_out, int out_size, void* d_ws, size_t ws_size,
                              hipStream_t stream) {
    (void)in_sizes; (void)n_in; (void)out_size; (void)ws_size;
    const float* x  = (const float*)d_in[0];
    const float* Wq = (const float*)d_in[1];
    const float* bq = (const float*)d_in[2];
    const float* Wk = (const float*)d_in[3];
    const float* bk = (const float*)d_in[4];
    const float* Wv = (const float*)d_in[5];
    const float* bv = (const float*)d_in[6];
    const float* Wo = (const float*)d_in[7];
    const float* bo = (const float*)d_in[8];

    const size_t M = (size_t)BATCH * SEQ;   // 8192
    const size_t E = EMB;

    char* ws = (char*)d_ws;
    bf16_t* Xb  = (bf16_t*)ws; ws += M * E * 2;
    bf16_t* Wqb = (bf16_t*)ws; ws += E * E * 2;
    bf16_t* Wkb = (bf16_t*)ws; ws += E * E * 2;
    bf16_t* Wvb = (bf16_t*)ws; ws += E * E * 2;
    bf16_t* Wob = (bf16_t*)ws; ws += E * E * 2;
    bf16_t* Qb  = (bf16_t*)ws; ws += M * E * 2;
    bf16_t* Kb  = (bf16_t*)ws; ws += M * E * 2;
    bf16_t* Vb  = (bf16_t*)ws; ws += M * E * 2;
    bf16_t* Vtb = (bf16_t*)ws; ws += M * E * 2;
    bf16_t* Ob  = (bf16_t*)ws; ws += M * E * 2;

    int nx = (int)(M * E);          // 8,388,608
    int nw = (int)(E * E);          // 1,048,576
    cvt_f32_bf16<<<(nx + 255) / 256, 256, 0, stream>>>(x,  Xb,  nx);
    cvt_f32_bf16<<<(nw + 255) / 256, 256, 0, stream>>>(Wq, Wqb, nw);
    cvt_f32_bf16<<<(nw + 255) / 256, 256, 0, stream>>>(Wk, Wkb, nw);
    cvt_f32_bf16<<<(nw + 255) / 256, 256, 0, stream>>>(Wv, Wvb, nw);
    cvt_f32_bf16<<<(nw + 255) / 256, 256, 0, stream>>>(Wo, Wob, nw);

    int gemm_blocks = (int)(((M / 32) * (E / 64) + 7) / 8);   // 512
    gemm_bias<true><<<gemm_blocks, 256, 0, stream>>>(Xb, Wqb, bq, Qb, (int)M, (int)E, (int)E);
    gemm_bias<true><<<gemm_blocks, 256, 0, stream>>>(Xb, Wkb, bk, Kb, (int)M, (int)E, (int)E);
    gemm_bias<true><<<gemm_blocks, 256, 0, stream>>>(Xb, Wvb, bv, Vb, (int)M, (int)E, (int)E);

    transpose_v<<<(nx + 255) / 256, 256, 0, stream>>>(Vb, Vtb, nx);

    int attn_blocks = (int)(BATCH * NHEAD * (SEQ / 16) / 8);  // 1024
    attn_fwd<<<attn_blocks, 256, 0, stream>>>(Qb, Kb, Vtb, Ob);

    gemm_bias<false><<<gemm_blocks, 256, 0, stream>>>(Ob, Wob, bo, d_out, (int)M, (int)E, (int)E);
}
